// BoltzmannMachine_19868518711450
// MI455X (gfx1250) — matrix operand order repaired and checked by reading, not compiled, not run on hardware
//
#include <hip/hip_runtime.h>
#include <hip/hip_bf16.h>
#include <cstdint>
#include <cstddef>

#define NTOT 8192
#define NVIS 4096
#define NHID 4096

typedef __attribute__((ext_vector_type(16))) _Float16 v16h;
typedef __attribute__((ext_vector_type(8)))  float    v8f;

// ---------------------------------------------------------------------------
// Kernel 1: compact the free (unclamped) steps of `perm` into a dense list.
// Single wave32; wave-level stream compaction with ballot/popc.
// ---------------------------------------------------------------------------
__global__ void bm_compact(const int* __restrict__ perm,
                           const int* __restrict__ clampv,
                           const float* __restrict__ u,
                           int* __restrict__ jfree,     // free index (j - NVIS), perm order
                           float* __restrict__ ufree) { // matching uniform draw
  const int lane = threadIdx.x & 31;
  int base = 0;
  for (int i = 0; i < NTOT; i += 32) {
    const int j = perm[i + lane];
    const bool fr = (clampv[j] == 0);
    const unsigned long long mask = __ballot(fr) & 0xffffffffull;
    const int pos = base + __popcll(mask & ((1ull << lane) - 1ull));
    if (fr) {
      jfree[pos] = j - NVIS;
      ufree[pos] = u[j];
    }
    base += __popcll(mask);
  }
}

// ---------------------------------------------------------------------------
// Kernel 2: initial local fields h[j] = W[j,:] . state  for free rows via WMMA.
// Grid: 256 blocks x 16 rows; 256 threads = 8 waves, each wave covers K=1024.
// B fragment holds `state` replicated across all 16 columns, so every column
// of D equals the row dot product. Memory-bound: WMMA compute waste is free.
// ---------------------------------------------------------------------------
__global__ void __launch_bounds__(256)
bm_fields_wmma(const float* __restrict__ W,
               const float* __restrict__ state,
               float* __restrict__ hbuf) {
  __shared__ __align__(32) _Float16 stateH[NTOT];
  __shared__ float red[8][16];

  const int tid = threadIdx.x;
  for (int i = tid; i < NTOT; i += 256) stateH[i] = (_Float16)state[i];
  __syncthreads();

  const int wave = tid >> 5;
  const int lane = tid & 31;
  const int m    = lane & 15;   // row within tile
  const int hsel = lane >> 4;   // lane-half -> K-group select
  const int row  = NVIS + blockIdx.x * 16 + m;
  const float* __restrict__ rowPtr = W + (size_t)row * NTOT;

  v8f c = {};
  const int k0 = wave * 1024;
#pragma unroll 2
  for (int kk = 0; kk < 1024; kk += 32) {
    const int kb = k0 + kk;
    // A fragment: 16x32 f16. element e -> K = hsel*16+e
    const float* p = rowPtr + kb + hsel * 16;
    const float4 f0 = *(const float4*)(p);
    const float4 f1 = *(const float4*)(p + 4);
    const float4 f2 = *(const float4*)(p + 8);
    const float4 f3 = *(const float4*)(p + 12);
    v16h a;
    a[0]=(_Float16)f0.x; a[1]=(_Float16)f0.y; a[2]=(_Float16)f0.z; a[3]=(_Float16)f0.w;
    a[4]=(_Float16)f1.x; a[5]=(_Float16)f1.y; a[6]=(_Float16)f1.z; a[7]=(_Float16)f1.w;
    a[8]=(_Float16)f2.x; a[9]=(_Float16)f2.y; a[10]=(_Float16)f2.z; a[11]=(_Float16)f2.w;
    a[12]=(_Float16)f3.x; a[13]=(_Float16)f3.y; a[14]=(_Float16)f3.z; a[15]=(_Float16)f3.w;
    // B fragment: 32x16 f16, all 16 columns identical = state chunk.
    // b[e] = state[kb + hsel*16 + e]
    const v16h b = *(const v16h*)&stateH[kb + hsel * 16];
    c = __builtin_amdgcn_wmma_f32_16x16x32_f16(false, a, false, b,
                                               (short)0, c, false, false);
  }

  // Column 0 of D: rows 0-7 live in lane 0 c[0..7], rows 8-15 in lane 16 c[0..7].
  if (lane == 0 || lane == 16) {
    const int rb = hsel * 8;
#pragma unroll
    for (int r = 0; r < 8; ++r) red[wave][rb + r] = c[r];
  }
  __syncthreads();
  if (tid < 16) {
    float s = 0.f;
#pragma unroll
    for (int w = 0; w < 8; ++w) s += red[w][tid];
    hbuf[blockIdx.x * 16 + tid] = s;
  }
}

// ---------------------------------------------------------------------------
// Kernel 3: sequential Gibbs sweep, blocked by 16 steps.  Single workgroup
// (1024 threads).  h lives in LDS; next block's 16x16 coupling matrix is
// gathered with global_load_async_to_lds_b32 (ASYNCcnt pipeline) while the
// current rank-16 update streams W rows from L2 (with global_prefetch ahead).
// ---------------------------------------------------------------------------
__global__ void __launch_bounds__(1024)
bm_gibbs(const float* __restrict__ W,
         const float* __restrict__ init_state,
         const float* __restrict__ Tptr,
         const int* __restrict__ jfreeG,
         const float* __restrict__ ufreeG,
         const float* __restrict__ hbuf,
         float* __restrict__ out) {
  __shared__ __align__(16) float h_s[NHID];     // local fields of free units
  __shared__ float    Mb[2][16][17];            // double-buffered 16x16 coupling
  __shared__ int      jb[2][16];
  __shared__ float    ub[2][16];
  __shared__ float    dvals[16];                // block flip deltas in {0,+-2}
  __shared__ unsigned sbits[NHID / 32];         // bipolar state: bit=1 -> +1

  const int tid = threadIdx.x;

  // ---- prologue ----
  {
    const float4 hv = *(const float4*)(hbuf + tid * 4);
    *(float4*)&h_s[tid * 4] = hv;
  }
  if (tid < NHID / 32) {
    unsigned w = 0u;
    for (int b = 0; b < 32; ++b)
      if (init_state[NVIS + tid * 32 + b] > 0.f) w |= (1u << b);
    sbits[tid] = w;
  }
  if (tid < 16)       jb[0][tid]      = jfreeG[tid];
  else if (tid < 32)  ub[0][tid - 16] = ufreeG[tid - 16];
  __syncthreads();
  if (tid < 256) {  // synchronous gather of block 0's coupling matrix
    const int t = tid >> 4, s = tid & 15;
    Mb[0][t][s] = W[(size_t)(NVIS + jb[0][t]) * NTOT + NVIS + jb[0][s]];
  }
  __syncthreads();

  const float Tv = Tptr[0];
  const int NB = NHID / 16;  // 256 macro-blocks

  for (int mb = 0; mb < NB; ++mb) {
    const int cur = mb & 1, nxt = cur ^ 1;
    const bool hasNext = (mb + 1 < NB);
    if (hasNext) {
      if (tid < 16)      jb[nxt][tid]      = jfreeG[(mb + 1) * 16 + tid];
      else if (tid < 32) ub[nxt][tid - 16] = ufreeG[(mb + 1) * 16 + (tid - 16)];
    }
    __syncthreads();  // (A) h updates + next-block metadata visible

    // ---- sequential flip resolution for 16 steps (one thread) ----
    if (tid == 0) {
#pragma unroll 1
      for (int t = 0; t < 16; ++t) {
        const int jj = jb[cur][t];
        float hh = h_s[jj];
        for (int s = 0; s < t; ++s) hh += dvals[s] * Mb[cur][t][s];
        const float p = 1.f / (1.f + __expf(-hh / Tv));
        const unsigned word = sbits[jj >> 5];
        const int oldb = (word >> (jj & 31)) & 1;
        const int newb = (ub[cur][t] <= p) ? 1 : 0;
        dvals[t] = 2.f * (float)(newb - oldb);
        if (newb != oldb) sbits[jj >> 5] = word ^ (1u << (jj & 31));
      }
    }
    __syncthreads();  // (B) dvals published

    // ---- async gather of NEXT block's 16x16 coupling matrix into LDS ----
    if (hasNext && tid < 256) {
      const int t = tid >> 4, s = tid & 15;
      const float* gp = W + (size_t)(NVIS + jb[nxt][t]) * NTOT + NVIS + jb[nxt][s];
      const unsigned ldsoff = (unsigned)(size_t)&Mb[nxt][t][s];
      asm volatile("global_load_async_to_lds_b32 %0, %1, off"
                   :: "v"(ldsoff), "v"(gp) : "memory");
    }

    // ---- rank-16 update of h (coalesced stream of W rows from L2) ----
    {
      const int k = tid * 4;
      float4 acc = *(const float4*)&h_s[k];
#pragma unroll 1
      for (int s = 0; s < 16; ++s) {
        const float d = dvals[s];       // uniform across block
        if (d != 0.f) {
          const float* rp = W + (size_t)(NVIS + jb[cur][s]) * NTOT + NVIS + k;
          const float4 w4 = *(const float4*)rp;
          acc.x += d * w4.x; acc.y += d * w4.y;
          acc.z += d * w4.z; acc.w += d * w4.w;
        }
      }
      *(float4*)&h_s[k] = acc;
    }

    // ---- prefetch next block's rows (16 rows x 16 KB, 64 threads/row) ----
    if (hasNext) {
      const int s = tid & 15;
      const int chunk = tid >> 4;  // 0..63 -> 256B strides cover 16KB
      const float* bp = W + (size_t)(NVIS + jb[nxt][s]) * NTOT + NVIS + chunk * 64;
      __builtin_prefetch(bp, 0, 1);
      __builtin_prefetch(bp + 32, 0, 1);
    }

    asm volatile("s_wait_asynccnt 0" ::: "memory");
    __syncthreads();  // (C) Mb[nxt] complete; h writes done
  }
  __syncthreads();

  // ---- epilogue: write outputs ----
  {
    const int k = tid * 4;
    const float4 iv = *(const float4*)(init_state + k);  // clamped prefix
    *(float4*)(out + k) = iv;
#pragma unroll
    for (int q = 0; q < 4; ++q) {
      const int jj = k + q;
      const int bit = (sbits[jj >> 5] >> (jj & 31)) & 1;
      out[NVIS + jj] = bit ? 1.f : -1.f;
    }
  }
}

// ---------------------------------------------------------------------------
extern "C" void kernel_launch(void* const* d_in, const int* in_sizes, int n_in,
                              void* d_out, int out_size, void* d_ws, size_t ws_size,
                              hipStream_t stream) {
  (void)in_sizes; (void)n_in; (void)out_size; (void)ws_size;
  const float* W      = (const float*)d_in[0];
  const float* st0    = (const float*)d_in[1];
  const float* u      = (const float*)d_in[2];
  const float* T      = (const float*)d_in[3];
  const int*   clampv = (const int*)d_in[4];
  const int*   perm   = (const int*)d_in[5];
  float* out = (float*)d_out;

  // workspace layout: [0,16K) h ; [16K,32K) jfree ; [32K,48K) ufree
  float* hbuf  = (float*)d_ws;
  int*   jfree = (int*)((char*)d_ws + 16 * 1024);
  float* ufree = (float*)((char*)d_ws + 32 * 1024);

  bm_compact<<<1, 32, 0, stream>>>(perm, clampv, u, jfree, ufree);
  bm_fields_wmma<<<256, 256, 0, stream>>>(W, st0, hbuf);
  bm_gibbs<<<1, 1024, 0, stream>>>(W, st0, T, jfree, ufree, hbuf, out);
}
